// MultiHeadAttention_55164559950124
// MI455X (gfx1250) — matrix operand order repaired and checked by reading, not compiled, not run on hardware
//
#include <hip/hip_runtime.h>
#include <hip/hip_bf16.h>

// ---------------------------------------------------------------------------
// MHA for MI455X (gfx1250, wave32, WMMA + Tensor Data Mover).
//
// Pipeline:
//   1) gemm_wt<A_F32, O_BF16_HEADS>   : Qh = q @ wq^T + bq   -> (b,h,s,dk) bf16
//   2) gemm_wt<A_F32, O_BF16_HEADS>   : Kh = k @ wk^T + bk   -> (b,h,s,dk) bf16
//   3) gemm_wt<A_F32, O_BF16_HEADS_T> : Vt = v @ wv^T + bv   -> (b,h,dk,s) bf16
//   4) attn_kernel (flash, causal)    : Ao  -> (b,h,s,dk) bf16
//   5) gemm_wt<A_BF16_HEADS, O_F32>   : out = Ao @ wo^T + bo -> (b,s,d) f32
//
// GEMM tiles are moved Global->LDS by TENSOR_LOAD_TO_LDS (TDM, TENSORcnt),
// all matmul math via v_wmma_f32_16x16x32_bf16 (f32 accumulate).
// ---------------------------------------------------------------------------

typedef __attribute__((ext_vector_type(16))) __bf16 v16bf;
typedef __attribute__((ext_vector_type(8)))  float  v8f;

constexpr int D_MODEL = 1024;
constexpr int NH      = 16;
constexpr int DK      = 64;
constexpr int BATCH   = 4;
constexpr int SEQ     = 2048;
constexpr int ROWS    = BATCH * SEQ;   // 8192

enum AMode { A_F32_ROWMAJOR, A_BF16_HEADS };
enum OMode { O_BF16_HEADS, O_BF16_HEADS_T, O_F32_ROWMAJOR };

// ---------------------------------------------------------------------------
// Tensor Data Mover: 2D tile load Global -> LDS.
// ---------------------------------------------------------------------------
#if defined(__HIP_DEVICE_COMPILE__) &&                           \
    __has_builtin(__builtin_amdgcn_tensor_load_to_lds) &&        \
    __has_builtin(__builtin_amdgcn_s_wait_tensorcnt)
#define USE_TDM 1
#else
#define USE_TDM 0
#endif

#if USE_TDM
typedef __attribute__((ext_vector_type(4))) unsigned int tdm_u32x4;
typedef __attribute__((ext_vector_type(8))) int          tdm_i32x8;
typedef __attribute__((ext_vector_type(4))) int          tdm_i32x4;

// data_size_code: 0=1B, 1=2B, 2=4B.  row_stride in elements (data_size units).
__device__ __forceinline__ void tdm_load_2d(unsigned lds_byte_addr,
                                            const void* gptr,
                                            unsigned data_size_code,
                                            unsigned tile_cols,
                                            unsigned tile_rows,
                                            unsigned long long row_stride) {
  const unsigned long long ga = (unsigned long long)(size_t)gptr;
  tdm_u32x4 g0;
  g0[0] = 1u;                                        // count=1 (valid user D#)
  g0[1] = lds_byte_addr;                             // LDS dest (bytes)
  g0[2] = (unsigned)ga;                              // global_addr[31:0]
  g0[3] = (unsigned)((ga >> 32) & 0x01ffffffu) | (2u << 30);  // [56:32], type=2
  // Group 1 bitfields (ISA 8.4): huge tensor dims (tiles are always in-bounds)
  const unsigned long long td0 = 0x7fffffffull, td1 = 0x7fffffffull;
  unsigned g[8];
  g[0] = (data_size_code & 3u) << 16;                // wg_mask=0, data_size
  g[1] = (unsigned)((td0 & 0xffffull) << 16);        // tensor_dim0[15:0]
  g[2] = (unsigned)((td0 >> 16) & 0xffffull) |       // tensor_dim0[31:16]
         (unsigned)((td1 & 0xffffull) << 16);        // tensor_dim1[15:0]
  g[3] = (unsigned)((td1 >> 16) & 0xffffull) |       // tensor_dim1[31:16]
         ((tile_cols & 0xffffu) << 16);              // tile_dim0
  g[4] = (tile_rows & 0xffffu);                      // tile_dim1 (tile_dim2=0)
  g[5] = (unsigned)(row_stride & 0xffffffffull);     // dim0_stride[31:0]
  g[6] = (unsigned)((row_stride >> 32) & 0xffffull); // dim0_stride[47:32]
  g[7] = 0u;                                         // dim1_stride=0 (2D)
  tdm_i32x8 g1 = {(int)g[0], (int)g[1], (int)g[2], (int)g[3],
                  (int)g[4], (int)g[5], (int)g[6], (int)g[7]};
  tdm_i32x4 z4 = {0, 0, 0, 0};
#if __clang_major__ >= 23
  tdm_i32x8 z8 = {0, 0, 0, 0, 0, 0, 0, 0};
  __builtin_amdgcn_tensor_load_to_lds(g0, g1, z4, z4, z8, 0);
#else
  __builtin_amdgcn_tensor_load_to_lds(g0, g1, z4, z4, 0);
#endif
}
#endif  // USE_TDM

// ---------------------------------------------------------------------------
// Generic  out = X @ W^T + bias  (M=8192, N=1024, K=1024)
// Block tile 64x64, 4 waves. Wave: 16 rows x 64 cols (4 f32 accumulators).
// Per K=32 step: TDM moves A(64x32) and W(64x32) tiles into LDS in source
// precision; fragments convert to bf16 during the LDS gather.
//
// WMMA fragment layouts used by this code (wave32; A/B share a contiguous k order, not the ISA table order):
//   A (16x32): lane (m=lane&15, hi=lane>>4); elem e -> K = e+hi*16
//   B (32x16): lane col n=lane&15;           elem e -> K = e+hi*16
//   C/D (16x16 f32): vgpr r -> row r+hi*8; col lane&15
// ---------------------------------------------------------------------------
template <AMode AM, OMode OM>
__global__ __launch_bounds__(128) void gemm_wt_kernel(
    const void* __restrict__ Aptr, const float* __restrict__ W,
    const float* __restrict__ bias, void* __restrict__ Out) {
  // [0, 8K):  A tile, 64x32 f32 (or 64x32 bf16 in head mode, 4K used)
  // [8K,16K): W tile, 64x32 f32
  __shared__ __align__(16) unsigned char smem[16384];
  float*  Asf = (float*)smem;
  __bf16* Asb = (__bf16*)smem;
  float*  Bsf = (float*)(smem + 8192);

  const int tid  = threadIdx.x;
  const int wave = tid >> 5;
  const int lane = tid & 31;
  const int nl   = lane & 15;
  const int hi   = lane >> 4;
  const int row0 = blockIdx.x * 64;
  const int col0 = blockIdx.y * 64;

  v8f acc[4] = {};

  for (int k0 = 0; k0 < D_MODEL; k0 += 32) {
#if USE_TDM
    if (wave == 0) {
      // ---- A tile via TDM ----
      if (AM == A_F32_ROWMAJOR) {
        tdm_load_2d((unsigned)(size_t)(void*)smem,
                    (const float*)Aptr + (size_t)row0 * D_MODEL + k0,
                    /*f32*/ 2u, 32u, 64u, (unsigned long long)D_MODEL);
      } else {
        // bf16 (b,h,s,dk): 64-row tile has fixed b (2048%64==0) and fixed h
        // (32-col slab stays inside one 64-wide head); row stride = DK elems.
        const int b = row0 >> 11, s0 = row0 & 2047;
        const int h = k0 >> 6, dk0 = k0 & 63;
        tdm_load_2d((unsigned)(size_t)(void*)smem,
                    (const __bf16*)Aptr +
                        (((size_t)b * NH + h) * SEQ + s0) * DK + dk0,
                    /*bf16*/ 1u, 32u, 64u, (unsigned long long)DK);
      }
      // ---- W tile via TDM ----
      tdm_load_2d((unsigned)(size_t)(void*)smem + 8192u,
                  W + (size_t)col0 * D_MODEL + k0, /*f32*/ 2u, 32u, 64u,
                  (unsigned long long)D_MODEL);
      __builtin_amdgcn_s_wait_tensorcnt(0);
    }
#else
    // ---- fallback: cooperative vector loads ----
    for (int i = tid; i < 64 * 32; i += 128) {
      const int r = i >> 5, c = i & 31;
      if (AM == A_F32_ROWMAJOR) {
        Asf[i] = ((const float*)Aptr)[(size_t)(row0 + r) * D_MODEL + (k0 + c)];
      } else {
        const int grow = row0 + r, gcol = k0 + c;
        const int b = grow >> 11, s = grow & 2047;
        const int h = gcol >> 6, dk = gcol & 63;
        Asb[i] =
            ((const __bf16*)Aptr)[(((size_t)b * NH + h) * SEQ + s) * DK + dk];
      }
      Bsf[i] = W[(size_t)(col0 + r) * D_MODEL + (k0 + c)];
    }
#endif
    __syncthreads();

    // ---- A fragment (convert to bf16 during gather) ----
    v16bf afrag;
#pragma unroll
    for (int e = 0; e < 16; ++e) {
      const int kk = e + hi * 16;
      const int idx = (wave * 16 + nl) * 32 + kk;
      afrag[e] = (AM == A_F32_ROWMAJOR) ? (__bf16)Asf[idx] : Asb[idx];
    }
#pragma unroll
    for (int nt = 0; nt < 4; ++nt) {
      v16bf bfrag;
#pragma unroll
      for (int e = 0; e < 16; ++e)
        bfrag[e] = (__bf16)Bsf[(nt * 16 + nl) * 32 + e + hi * 16];
      acc[nt] = __builtin_amdgcn_wmma_f32_16x16x32_bf16(
          false, afrag, false, bfrag, (short)0, acc[nt], false, false);
    }
    __syncthreads();
  }

  // ---- epilogue: add bias, scatter per output layout ----
#pragma unroll
  for (int nt = 0; nt < 4; ++nt) {
    const int col = col0 + nt * 16 + nl;
    const float bv = bias[col];
#pragma unroll
    for (int r = 0; r < 8; ++r) {
      const int row = row0 + wave * 16 + r + hi * 8;
      const float val = acc[nt][r] + bv;
      if (OM == O_F32_ROWMAJOR) {
        ((float*)Out)[(size_t)row * D_MODEL + col] = val;
      } else {
        const int b = row >> 11, s = row & 2047;
        const int h = col >> 6, dk = col & 63;
        if (OM == O_BF16_HEADS)
          ((__bf16*)Out)[(((size_t)b * NH + h) * SEQ + s) * DK + dk] =
              (__bf16)val;
        else  // O_BF16_HEADS_T: (b,h,dk,s) -> P*V B-fragments load contiguously
          ((__bf16*)Out)[(((size_t)b * NH + h) * DK + dk) * SEQ + s] =
              (__bf16)val;
      }
    }
  }
}

// ---------------------------------------------------------------------------
// Causal flash attention: one wave per (b*h, 16-query tile).
// Streams 32-key tiles; scores + P*V via v_wmma_f32_16x16x32_bf16.
// Row softmax stats reduced with shfl_xor within each 16-lane half.
// ---------------------------------------------------------------------------
__global__ __launch_bounds__(32) void attn_kernel(
    const __bf16* __restrict__ Qh, const __bf16* __restrict__ Kh,
    const __bf16* __restrict__ Vt, __bf16* __restrict__ Outh) {
  const int bh   = blockIdx.x >> 7;   // b*NH + h
  const int qb   = blockIdx.x & 127;  // query tile
  const int lane = threadIdx.x;
  const int nl   = lane & 15;
  const int hi   = lane >> 4;
  const int q0   = qb * 16;

  const __bf16* Qp = Qh + (size_t)bh * SEQ * DK;
  const __bf16* Kp = Kh + (size_t)bh * SEQ * DK;
  const __bf16* Vp = Vt + (size_t)bh * DK * SEQ;

  // Q A-fragments, 2 chunks over d (contiguous 8-element groups)
  v16bf qf[2];
  {
    const __bf16* qrow = Qp + (size_t)(q0 + nl) * DK;
#pragma unroll
    for (int c = 0; c < 2; ++c) {
#pragma unroll
      for (int e = 0; e < 8; ++e) qf[c][e] = qrow[c * 32 + hi * 16 + e];
#pragma unroll
      for (int e = 0; e < 8; ++e) qf[c][8 + e] = qrow[c * 32 + hi * 16 + 8 + e];
    }
  }

  v8f oacc[4] = {};  // rows r+hi*8, d-cols nt*16+nl
  float mrow[8], lrow[8];
#pragma unroll
  for (int r = 0; r < 8; ++r) { mrow[r] = -1e30f; lrow[r] = 0.0f; }

  __shared__ __bf16 Pbuf[16][40];  // 16x32 P tile (padded) for D->A relayout

  const int nk = ((q0 + 15) >> 5) + 1;  // 32-key tiles under causal mask
  const float scale = 0.125f;           // 1/sqrt(64)

  for (int kb = 0; kb < nk; ++kb) {
    const int kbase = kb * 32;

    // ---- scores: S[16x32] = Q(16x64) x K^T, two 16x16 D frags ----
    v8f s[2] = {};
#pragma unroll
    for (int half = 0; half < 2; ++half) {
#pragma unroll
      for (int c = 0; c < 2; ++c) {
        v16bf kfrag;
        const __bf16* krow =
            Kp + (size_t)(kbase + half * 16 + nl) * DK + c * 32 + hi * 16;
#pragma unroll
        for (int e = 0; e < 16; ++e) kfrag[e] = krow[e];
        s[half] = __builtin_amdgcn_wmma_f32_16x16x32_bf16(
            false, qf[c], false, kfrag, (short)0, s[half], false, false);
      }
    }

    // ---- scale + causal mask + online softmax update ----
#pragma unroll
    for (int r = 0; r < 8; ++r) {
      const int qi = q0 + r + hi * 8;
#pragma unroll
      for (int half = 0; half < 2; ++half) {
        const int ki = kbase + half * 16 + nl;
        const float v = s[half][r] * scale;
        s[half][r] = (ki <= qi) ? v : -1e30f;
      }
      float mx = fmaxf(s[0][r], s[1][r]);
      mx = fmaxf(mx, __shfl_xor(mx, 1, 32));
      mx = fmaxf(mx, __shfl_xor(mx, 2, 32));
      mx = fmaxf(mx, __shfl_xor(mx, 4, 32));
      mx = fmaxf(mx, __shfl_xor(mx, 8, 32));
      const float mnew = fmaxf(mrow[r], mx);

      const float p0 = __expf(s[0][r] - mnew);
      const float p1 = __expf(s[1][r] - mnew);
      s[0][r] = p0;
      s[1][r] = p1;
      float rs = p0 + p1;
      rs += __shfl_xor(rs, 1, 32);
      rs += __shfl_xor(rs, 2, 32);
      rs += __shfl_xor(rs, 4, 32);
      rs += __shfl_xor(rs, 8, 32);

      const float alpha = __expf(mrow[r] - mnew);
      lrow[r] = lrow[r] * alpha + rs;
      mrow[r] = mnew;
#pragma unroll
      for (int nt = 0; nt < 4; ++nt) oacc[nt][r] *= alpha;
    }

    // ---- P: D-layout -> A-layout via LDS ----
#pragma unroll
    for (int half = 0; half < 2; ++half)
#pragma unroll
      for (int r = 0; r < 8; ++r)
        Pbuf[r + hi * 8][half * 16 + nl] = (__bf16)s[half][r];
    __syncthreads();
    v16bf pfrag;
    {
      const __bf16* prow = &Pbuf[nl][0];
#pragma unroll
      for (int e = 0; e < 8; ++e) pfrag[e] = prow[hi * 16 + e];
#pragma unroll
      for (int e = 0; e < 8; ++e) pfrag[8 + e] = prow[hi * 16 + 8 + e];
    }
    __syncthreads();

    // ---- O += P(16x32) x V(32x64); Vt layout -> contiguous B frags ----
#pragma unroll
    for (int nt = 0; nt < 4; ++nt) {
      v16bf vfrag;
      const __bf16* vrow = Vp + (size_t)(nt * 16 + nl) * SEQ + kbase + hi * 16;
#pragma unroll
      for (int e = 0; e < 16; ++e) vfrag[e] = vrow[e];
      oacc[nt] = __builtin_amdgcn_wmma_f32_16x16x32_bf16(
          false, pfrag, false, vfrag, (short)0, oacc[nt], false, false);
    }
  }

  // ---- finalize: /l, store bf16 (b,h,s,dk) ----
#pragma unroll
  for (int nt = 0; nt < 4; ++nt) {
#pragma unroll
    for (int r = 0; r < 8; ++r) {
      const int srow = q0 + r + hi * 8;
      const float val = oacc[nt][r] / lrow[r];
      Outh[((size_t)bh * SEQ + srow) * DK + nt * 16 + nl] = (__bf16)val;
    }
  }
}

// ---------------------------------------------------------------------------
extern "C" void kernel_launch(void* const* d_in, const int* in_sizes, int n_in,
                              void* d_out, int out_size, void* d_ws,
                              size_t ws_size, hipStream_t stream) {
  const float* q  = (const float*)d_in[0];
  const float* k  = (const float*)d_in[1];
  const float* v  = (const float*)d_in[2];
  // d_in[3] = mask (causal tril) -- applied analytically in attn_kernel
  const float* wq = (const float*)d_in[4];
  const float* bq = (const float*)d_in[5];
  const float* wk = (const float*)d_in[6];
  const float* bk = (const float*)d_in[7];
  const float* wv = (const float*)d_in[8];
  const float* bv = (const float*)d_in[9];
  const float* wo = (const float*)d_in[10];
  const float* bo = (const float*)d_in[11];

  char* ws = (char*)d_ws;
  __bf16* Qh = (__bf16*)(ws + ((size_t)0 << 20));   // 16 MiB each
  __bf16* Kh = (__bf16*)(ws + ((size_t)16 << 20));
  __bf16* Vt = (__bf16*)(ws + ((size_t)32 << 20));
  __bf16* Ao = (__bf16*)(ws + ((size_t)48 << 20));

  dim3 grid(ROWS / 64, D_MODEL / 64);
  dim3 blk(128);

  gemm_wt_kernel<A_F32_ROWMAJOR, O_BF16_HEADS>
      <<<grid, blk, 0, stream>>>(q, wq, bq, Qh);
  gemm_wt_kernel<A_F32_ROWMAJOR, O_BF16_HEADS>
      <<<grid, blk, 0, stream>>>(k, wk, bk, Kh);
  gemm_wt_kernel<A_F32_ROWMAJOR, O_BF16_HEADS_T>
      <<<grid, blk, 0, stream>>>(v, wv, bv, Vt);

  attn_kernel<<<dim3(BATCH * NH * (SEQ / 16)), dim3(32), 0, stream>>>(Qh, Kh,
                                                                      Vt, Ao);

  gemm_wt_kernel<A_BF16_HEADS, O_F32_ROWMAJOR>
      <<<grid, blk, 0, stream>>>(Ao, wo, bo, (float*)d_out);
}
